// MultiHeadAttention_41420664603202
// MI455X (gfx1250) — compile-verified, matrix operand order checked by reading, not run on hardware
//
#include <hip/hip_runtime.h>
#include <hip/hip_bf16.h>
#include <math.h>

#define B_  2
#define S_  2048
#define DM_ 1024
#define H_  16
#define HD_ 64

typedef __attribute__((ext_vector_type(16))) __bf16 v16bf;
typedef __attribute__((ext_vector_type(8)))  __bf16 bf16x8;
typedef __attribute__((ext_vector_type(8)))  float  v8f;

// ---- fp32 -> bf16 conversion: hardware packed cvt when available ----
__device__ __forceinline__ __bf16 f2bf(float f) { return (__bf16)f; }

__device__ __forceinline__ bf16x8 cvt8_bf16(float4 f0, float4 f1) {
  bf16x8 r;
#if __has_builtin(__builtin_amdgcn_cvt_pk_bf16_f32)
  typedef __attribute__((ext_vector_type(2))) __bf16 bf16x2;
  bf16x2 p0 = __builtin_amdgcn_cvt_pk_bf16_f32(f0.x, f0.y);
  bf16x2 p1 = __builtin_amdgcn_cvt_pk_bf16_f32(f0.z, f0.w);
  bf16x2 p2 = __builtin_amdgcn_cvt_pk_bf16_f32(f1.x, f1.y);
  bf16x2 p3 = __builtin_amdgcn_cvt_pk_bf16_f32(f1.z, f1.w);
  r[0] = p0[0]; r[1] = p0[1]; r[2] = p1[0]; r[3] = p1[1];
  r[4] = p2[0]; r[5] = p2[1]; r[6] = p3[0]; r[7] = p3[1];
#else
  r[0] = (__bf16)f0.x; r[1] = (__bf16)f0.y; r[2] = (__bf16)f0.z; r[3] = (__bf16)f0.w;
  r[4] = (__bf16)f1.x; r[5] = (__bf16)f1.y; r[6] = (__bf16)f1.z; r[7] = (__bf16)f1.w;
#endif
  return r;
}

__device__ __forceinline__ v8f wmma_bf16(v16bf a, v16bf b, v8f c) {
  // D = A(16x32) * B(32x16) + C, f32 accumulate
  return __builtin_amdgcn_wmma_f32_16x16x32_bf16(false, a, false, b, (short)0, c,
                                                 false, false);
}

// Load one WMMA 16-bit operand chunk (32 consecutive contraction elements at p)
// in CDNA5 register layout: lanes 0-15 hold K={0..7,16..23},
// lanes 16-31 hold K={8..15,24..31}.  p must be 16B aligned.
__device__ __forceinline__ v16bf load_op_bf16(const __bf16* p) {
  const int off = ((threadIdx.x & 31) >> 4) ? 8 : 0;
  bf16x8 lo = *(const bf16x8*)(p + off);
  bf16x8 hi = *(const bf16x8*)(p + off + 16);
  v16bf r;
  #pragma unroll
  for (int j = 0; j < 8; ++j) { r[j] = lo[j]; r[j + 8] = hi[j]; }
  return r;
}

// =====================================================================
// Kernel 0: one-shot fp32 -> bf16 conversion of activations + weights.
// Keeps all per-element cvt VALU work OUT of the GEMM inner loops.
// =====================================================================
__global__ __launch_bounds__(256)
void mha_cvt_bf16(const float* __restrict__ q, const float* __restrict__ k,
                  const float* __restrict__ v, const float* __restrict__ wq,
                  const float* __restrict__ wk, const float* __restrict__ wv,
                  const float* __restrict__ wo, __bf16* __restrict__ qb,
                  __bf16* __restrict__ kb, __bf16* __restrict__ vb,
                  __bf16* __restrict__ wqb, __bf16* __restrict__ wkb,
                  __bf16* __restrict__ wvb, __bf16* __restrict__ wob) {
  const int a = blockIdx.y;                     // which array
  const float* src; __bf16* dst; size_t n;
  switch (a) {
    case 0: src = q;  dst = qb;  n = (size_t)B_ * S_ * DM_; break;
    case 1: src = k;  dst = kb;  n = (size_t)B_ * S_ * DM_; break;
    case 2: src = v;  dst = vb;  n = (size_t)B_ * S_ * DM_; break;
    case 3: src = wq; dst = wqb; n = (size_t)DM_ * DM_;     break;
    case 4: src = wk; dst = wkb; n = (size_t)DM_ * DM_;     break;
    case 5: src = wv; dst = wvb; n = (size_t)DM_ * DM_;     break;
    default: src = wo; dst = wob; n = (size_t)DM_ * DM_;    break;
  }
  const size_t i = ((size_t)blockIdx.x * 256 + threadIdx.x) * 8;
  if (i >= n) return;
  float4 f0 = *(const float4*)(src + i);
  float4 f1 = *(const float4*)(src + i + 4);
  *(bf16x8*)(dst + i) = cvt8_bf16(f0, f1);
}

// =====================================================================
// Kernel 1: fused Q/K/V projection  y = x @ W^T  (W is [out,in]) — pure bf16.
// One wave computes a 32x64 tile: 8 WMMAs per k-step vs 6 operand loads.
// sel 0/1 -> (B,H,S,HD) bf16 row-major; sel 2 (V) -> (B,H,HD,S) transposed.
// =====================================================================
__global__ __launch_bounds__(32)
void mha_qkv_proj(const __bf16* __restrict__ qb, const __bf16* __restrict__ kb,
                  const __bf16* __restrict__ vb, const __bf16* __restrict__ wqb,
                  const __bf16* __restrict__ wkb, const __bf16* __restrict__ wvb,
                  __bf16* __restrict__ Qp, __bf16* __restrict__ Kp,
                  __bf16* __restrict__ Vt) {
  const int lane = threadIdx.x & 31;
  const int sel  = blockIdx.z;                    // 0=Q 1=K 2=V
  const int r0   = blockIdx.x * 32;               // row in flattened (B*S)
  const int h    = blockIdx.y;                    // head = 64-col group
  const int n0   = h * HD_;
  const __bf16* X = (sel == 0) ? qb : (sel == 1) ? kb : vb;
  const __bf16* W = (sel == 0) ? wqb : (sel == 1) ? wkb : wvb;
  const int nl = lane & 15;

  const __bf16* arow0 = X + (size_t)(r0 + nl) * DM_;
  const __bf16* arow1 = X + (size_t)(r0 + 16 + nl) * DM_;
  v8f acc[2][4] = {};
  for (int k0 = 0; k0 < DM_; k0 += 32) {
    __builtin_prefetch(arow0 + k0 + 64, 0, 0);    // global_prefetch_b8
    v16bf a0 = load_op_bf16(arow0 + k0);
    v16bf a1 = load_op_bf16(arow1 + k0);
    #pragma unroll
    for (int t = 0; t < 4; ++t) {
      v16bf bop = load_op_bf16(W + (size_t)(n0 + t * 16 + nl) * DM_ + k0);
      acc[0][t] = wmma_bf16(a0, bop, acc[0][t]);
      acc[1][t] = wmma_bf16(a1, bop, acc[1][t]);
    }
  }

  const int half = lane >> 4;                     // C layout: lanes 0-15 M=0..7
  const int b    = r0 / S_;
  const int s0   = r0 % S_;
  if (sel < 2) {
    __bf16* dst = ((sel == 0) ? Qp : Kp) + (size_t)(b * H_ + h) * S_ * HD_;
    #pragma unroll
    for (int g = 0; g < 2; ++g)
      #pragma unroll
      for (int t = 0; t < 4; ++t)
        #pragma unroll
        for (int i = 0; i < 8; ++i)
          dst[(size_t)(s0 + g * 16 + i + 8 * half) * HD_ + t * 16 + nl] =
              f2bf(acc[g][t][i]);
  } else {                                        // V stored transposed (HD,S)
    __bf16* dst = Vt + (size_t)(b * H_ + h) * HD_ * S_;
    #pragma unroll
    for (int g = 0; g < 2; ++g)
      #pragma unroll
      for (int t = 0; t < 4; ++t)
        #pragma unroll
        for (int i = 0; i < 8; ++i)
          dst[(size_t)(t * 16 + nl) * S_ + s0 + g * 16 + i + 8 * half] =
              f2bf(acc[g][t][i]);
  }
}

// =====================================================================
// Kernel 2: causal flash attention, one wave per (b, h, 16-query tile).
// QK^T: 2 bf16 WMMAs per 16-key tile (contraction d=64).
// Online softmax in f32; P transposed C-layout->A-layout through 1KB LDS
// (same-wave DS ops are in-order, no barrier needed).
// PV: 4 bf16 WMMAs per 32-key block against transposed V.
// =====================================================================
__global__ __launch_bounds__(32)
void mha_attn(const __bf16* __restrict__ Qp, const __bf16* __restrict__ Kp,
              const __bf16* __restrict__ Vt, __bf16* __restrict__ Ao) {
  __shared__ __align__(16) __bf16 ldsP[16 * 32];  // P tile, [m][key]
  const int lane = threadIdx.x & 31;
  const int q0 = blockIdx.x * 16;
  const int h  = blockIdx.y;
  const int b  = blockIdx.z;
  const int nl = lane & 15;
  const int half = lane >> 4;

  const __bf16* Qh = Qp + (size_t)(b * H_ + h) * S_ * HD_;
  const __bf16* Kh = Kp + (size_t)(b * H_ + h) * S_ * HD_;
  const __bf16* Vh = Vt + (size_t)(b * H_ + h) * HD_ * S_;

  // Query A-operands for both d-halves, loaded once.
  const v16bf qa0 = load_op_bf16(Qh + (size_t)(q0 + nl) * HD_ + 0);
  const v16bf qa1 = load_op_bf16(Qh + (size_t)(q0 + nl) * HD_ + 32);

  v8f o[4] = {};
  float m_i[8], l_i[8];
  #pragma unroll
  for (int i = 0; i < 8; ++i) { m_i[i] = -1.0e30f; l_i[i] = 0.0f; }

  const float scale = 0.125f;                     // 1/sqrt(64)
  for (int kb = 0; kb < q0 + 16; kb += 32) {      // causal: never past diagonal
    // ---- scores for keys [kb, kb+32) : two 16x16 N-tiles ----
    v8f s0 = v8f{}, s1 = v8f{};
    {
      v16bf kA = load_op_bf16(Kh + (size_t)(kb + nl) * HD_ + 0);
      v16bf kB = load_op_bf16(Kh + (size_t)(kb + nl) * HD_ + 32);
      s0 = wmma_bf16(qa0, kA, s0);
      s0 = wmma_bf16(qa1, kB, s0);
      kA = load_op_bf16(Kh + (size_t)(kb + 16 + nl) * HD_ + 0);
      kB = load_op_bf16(Kh + (size_t)(kb + 16 + nl) * HD_ + 32);
      s1 = wmma_bf16(qa0, kA, s1);
      s1 = wmma_bf16(qa1, kB, s1);
    }

    const bool needMask = (kb + 31 > q0);
    #pragma unroll
    for (int i = 0; i < 8; ++i) {
      const int qg = q0 + i + 8 * half;           // global query row
      float v0 = s0[i] * scale;
      float v1 = s1[i] * scale;
      if (needMask) {
        if (kb + nl      > qg) v0 = -1.0e30f;
        if (kb + 16 + nl > qg) v1 = -1.0e30f;
      }
      // row max across the 16 lanes of this half (xor offsets <=8 stay in-half)
      float mx = fmaxf(v0, v1);
      mx = fmaxf(mx, __shfl_xor(mx, 8, 32));
      mx = fmaxf(mx, __shfl_xor(mx, 4, 32));
      mx = fmaxf(mx, __shfl_xor(mx, 2, 32));
      mx = fmaxf(mx, __shfl_xor(mx, 1, 32));
      const float mnew = fmaxf(m_i[i], mx);
      const float corr = __expf(m_i[i] - mnew);
      m_i[i] = mnew;
      const float e0 = __expf(v0 - mnew);
      const float e1 = __expf(v1 - mnew);
      l_i[i] = l_i[i] * corr + e0 + e1;           // per-lane partial row sum
      #pragma unroll
      for (int t = 0; t < 4; ++t) o[t][i] = o[t][i] * corr;
      const int m = i + 8 * half;
      ldsP[m * 32 + nl]      = f2bf(e0);          // C-layout -> LDS [m][key]
      ldsP[m * 32 + 16 + nl] = f2bf(e1);
    }

    // Reload P in A-operand layout (same-wave LDS is in-order).
    const v16bf pa = load_op_bf16(&ldsP[nl * 32]);
    #pragma unroll
    for (int t = 0; t < 4; ++t) {                 // B operand: Vt[d][key], contiguous
      v16bf vb = load_op_bf16(Vh + (size_t)(t * 16 + nl) * S_ + kb);
      o[t] = wmma_bf16(pa, vb, o[t]);
    }
  }

  // ---- finalize: reduce row sums across the half, normalize, store bf16 ----
  #pragma unroll
  for (int i = 0; i < 8; ++i) {
    float l = l_i[i];
    l += __shfl_xor(l, 8, 32);
    l += __shfl_xor(l, 4, 32);
    l += __shfl_xor(l, 2, 32);
    l += __shfl_xor(l, 1, 32);
    const float inv = 1.0f / l;
    const int m = i + 8 * half;
    __bf16* row = Ao + ((size_t)b * S_ + q0 + m) * DM_ + h * HD_;
    #pragma unroll
    for (int t = 0; t < 4; ++t)
      row[t * 16 + nl] = f2bf(o[t][i] * inv);
  }
}

// =====================================================================
// Kernel 3: output projection  out = attn @ wo^T, fp32 result — pure bf16 in.
// One wave computes a 32x64 tile.
// =====================================================================
__global__ __launch_bounds__(32)
void mha_out_proj(const __bf16* __restrict__ Ao, const __bf16* __restrict__ wob,
                  float* __restrict__ out) {
  const int lane = threadIdx.x & 31;
  const int r0 = blockIdx.x * 32;
  const int n0 = blockIdx.y * 64;
  const int nl = lane & 15;

  const __bf16* arow0 = Ao + (size_t)(r0 + nl) * DM_;
  const __bf16* arow1 = Ao + (size_t)(r0 + 16 + nl) * DM_;
  v8f acc[2][4] = {};
  for (int k0 = 0; k0 < DM_; k0 += 32) {
    v16bf a0 = load_op_bf16(arow0 + k0);
    v16bf a1 = load_op_bf16(arow1 + k0);
    #pragma unroll
    for (int t = 0; t < 4; ++t) {
      v16bf bop = load_op_bf16(wob + (size_t)(n0 + t * 16 + nl) * DM_ + k0);
      acc[0][t] = wmma_bf16(a0, bop, acc[0][t]);
      acc[1][t] = wmma_bf16(a1, bop, acc[1][t]);
    }
  }
  const int half = lane >> 4;
  #pragma unroll
  for (int g = 0; g < 2; ++g)
    #pragma unroll
    for (int t = 0; t < 4; ++t)
      #pragma unroll
      for (int i = 0; i < 8; ++i)
        out[(size_t)(r0 + g * 16 + i + 8 * half) * DM_ + n0 + t * 16 + nl] =
            acc[g][t][i];
}

// =====================================================================
extern "C" void kernel_launch(void* const* d_in, const int* in_sizes, int n_in,
                              void* d_out, int out_size, void* d_ws, size_t ws_size,
                              hipStream_t stream) {
  (void)in_sizes; (void)n_in; (void)out_size; (void)ws_size;
  const float* q  = (const float*)d_in[0];
  const float* k  = (const float*)d_in[1];
  const float* v  = (const float*)d_in[2];
  const float* wq = (const float*)d_in[3];
  const float* wk = (const float*)d_in[4];
  const float* wv = (const float*)d_in[5];
  const float* wo = (const float*)d_in[6];

  // Workspace layout (bf16 segments):
  //   Qp | Kp | Vt | Ao   : B*S*DM each (8 MB each)
  //   qb | kb | vb        : B*S*DM each (8 MB each)
  //   wqb | wkb | wvb | wob : DM*DM each (2 MB each)      total 64 MB
  char* ws = (char*)d_ws;
  const size_t segX = (size_t)B_ * S_ * DM_ * sizeof(__bf16);
  const size_t segW = (size_t)DM_ * DM_ * sizeof(__bf16);
  __bf16* Qp  = (__bf16*)(ws);
  __bf16* Kp  = (__bf16*)(ws + 1 * segX);
  __bf16* Vt  = (__bf16*)(ws + 2 * segX);
  __bf16* Ao  = (__bf16*)(ws + 3 * segX);
  __bf16* qb  = (__bf16*)(ws + 4 * segX);
  __bf16* kb  = (__bf16*)(ws + 5 * segX);
  __bf16* vb  = (__bf16*)(ws + 6 * segX);
  __bf16* wqb = (__bf16*)(ws + 7 * segX);
  __bf16* wkb = (__bf16*)(ws + 7 * segX + 1 * segW);
  __bf16* wvb = (__bf16*)(ws + 7 * segX + 2 * segW);
  __bf16* wob = (__bf16*)(ws + 7 * segX + 3 * segW);

  // 0) one-shot fp32 -> bf16 conversion (activations + weights)
  {
    const size_t nmax = (size_t)B_ * S_ * DM_;            // largest array
    const int blocks = (int)(nmax / (8 * 256));           // 2048
    mha_cvt_bf16<<<dim3(blocks, 7), dim3(256), 0, stream>>>(
        q, k, v, wq, wk, wv, wo, qb, kb, vb, wqb, wkb, wvb, wob);
  }
  // 1) Q/K/V projections (bf16 WMMA)
  mha_qkv_proj<<<dim3((B_ * S_) / 32, H_, 3), dim3(32), 0, stream>>>(
      qb, kb, vb, wqb, wkb, wvb, Qp, Kp, Vt);
  // 2) causal flash attention (bf16 WMMA, f32 softmax)
  mha_attn<<<dim3(S_ / 16, H_, B_), dim3(32), 0, stream>>>(Qp, Kp, Vt, Ao);
  // 3) output projection (bf16 WMMA, f32 out)
  mha_out_proj<<<dim3((B_ * S_) / 32, DM_ / 64), dim3(32), 0, stream>>>(
      Ao, wob, (float*)d_out);
}
